// Trainer_91216515433187
// MI455X (gfx1250) — matrix operand order repaired and checked by reading, not compiled, not run on hardware
//
#include <hip/hip_runtime.h>
#include <hip/hip_bf16.h>
#include <stdint.h>

// ---------------------------------------------------------------------------
// Types for CDNA5 WMMA (wave32): v_wmma_f32_16x16x32_bf16
// ---------------------------------------------------------------------------
typedef __bf16 bf16x16 __attribute__((ext_vector_type(16)));
typedef float  f32x8   __attribute__((ext_vector_type(8)));

#define EPS 1e-3f
#define T_STEPS 1024
#define DEC_STEPS 35
#define KDIM 288          // 8 input slots + 256 h slots + 24 zero pad
#define KBLK 9            // 288 / 32
#define NDIM 1024         // 4*H gate columns
#define NTILES 64         // 1024 / 16
#define TILE_ELEMS 512    // 32 lanes * 16 bf16 per WMMA B tile

static __device__ __forceinline__ unsigned short f2bf(float x) {
    unsigned u = __float_as_uint(x);
    u += 0x7fffu + ((u >> 16) & 1u);        // round-to-nearest-even
    return (unsigned short)(u >> 16);
}
static __device__ __forceinline__ float sigm(float x) {
    return 1.0f / (1.0f + __expf(-x));
}
static __device__ __forceinline__ float tanh_f(float x) {
    return 2.0f / (1.0f + __expf(-2.0f * x)) - 1.0f;
}

// ---------------------------------------------------------------------------
// Setup kernel 1: fold BN+Dense+lstm_k into B_enc rows 0..7, W_eis@lstm_k into
// B_dec rows 0..1, lstm_rk into rows 8..263, convert to bf16 and swizzle into
// the contiguous-k B-operand lane layout this file uses (A_lds matches):
//   lane = (n%16) + 16*(K%32 >= 16),  elem = K%16, tile-major [kblk][ntile]
// ---------------------------------------------------------------------------
__global__ void pack_B_kernel(const float* __restrict__ W_pulse,
                              const float* __restrict__ W_eis,
                              const float* __restrict__ lstm_k,
                              const float* __restrict__ lstm_rk,
                              const float* __restrict__ bn_gamma,
                              const float* __restrict__ bn_var,
                              unsigned short* __restrict__ bpk_enc,
                              unsigned short* __restrict__ bpk_dec) {
    const int PER = KDIM * NDIM;
    int idx = blockIdx.x * blockDim.x + threadIdx.x;
    if (idx >= 2 * PER) return;
    int which = idx / PER;
    int rem   = idx - which * PER;
    int k = rem >> 10;
    int n = rem & 1023;

    float val = 0.0f;
    if (k >= 8 && k < 264) {
        val = lstm_rk[(size_t)(k - 8) * 1024 + n];
    } else if (which == 0 && k < 8) {
        float s = bn_gamma[k] * rsqrtf(bn_var[k] + EPS);
        float acc = 0.0f;
        for (int h = 0; h < 256; ++h)
            acc += W_pulse[k * 256 + h] * lstm_k[(size_t)h * 1024 + n];
        val = s * acc;
    } else if (which == 1 && k < 2) {
        float acc = 0.0f;
        for (int h = 0; h < 256; ++h)
            acc += W_eis[k * 256 + h] * lstm_k[(size_t)h * 1024 + n];
        val = acc;
    }

    int kb   = k >> 5;
    int kk   = k & 31;
    int lane = (n & 15) + ((kk >> 4) << 4);
    int e    = kk & 15;
    int nt   = n >> 4;
    unsigned short* dst = which ? bpk_dec : bpk_enc;
    dst[((((kb * NTILES) + nt) * 32 + lane) << 4) + e] = f2bf(val);
}

// ---------------------------------------------------------------------------
// Setup kernel 2: folded bias vectors (length 1024 each)
// ---------------------------------------------------------------------------
__global__ void bias_kernel(const float* __restrict__ bn_gamma,
                            const float* __restrict__ bn_beta,
                            const float* __restrict__ bn_mean,
                            const float* __restrict__ bn_var,
                            const float* __restrict__ W_pulse,
                            const float* __restrict__ b_pulse,
                            const float* __restrict__ lstm_k,
                            const float* __restrict__ lstm_b,
                            const float* __restrict__ b_eis,
                            float* __restrict__ bias_enc,
                            float* __restrict__ bias_dec) {
    int idx = blockIdx.x * blockDim.x + threadIdx.x;
    if (idx >= 2048) return;
    int which = idx >> 10;
    int n     = idx & 1023;
    float acc = lstm_b[n];
    if (which == 0) {
        for (int h = 0; h < 256; ++h) {
            float sw = b_pulse[h];
            for (int f = 0; f < 8; ++f) {
                float s = bn_gamma[f] * rsqrtf(bn_var[f] + EPS);
                sw += (bn_beta[f] - bn_mean[f] * s) * W_pulse[f * 256 + h];
            }
            acc += sw * lstm_k[(size_t)h * 1024 + n];
        }
        bias_enc[n] = acc;
    } else {
        for (int h = 0; h < 256; ++h)
            acc += b_eis[h] * lstm_k[(size_t)h * 1024 + n];
        bias_dec[n] = acc;
    }
}

// ---------------------------------------------------------------------------
// Main persistent LSTM kernel: 16 blocks x 256 threads (8 wave32).
// Each block owns 16 batch rows for the entire sequence.
// Wave w owns gate-aligned N-tiles {2w+p, 2w+p+16, +32, +48} (p=0,1), so the
// i/f/g/o pre-activations for one column live in the same lane -> gate math
// runs entirely in WMMA accumulator registers (no z staging in LDS).
// ---------------------------------------------------------------------------
__global__ __launch_bounds__(256)
void lstm_kernel(const float* __restrict__ pulse,
                 const float* __restrict__ embed_table,
                 const float* __restrict__ W_mlp,
                 const float* __restrict__ b_mlp,
                 const float* __restrict__ scale_w,
                 const float* __restrict__ scale_b,
                 const unsigned short* __restrict__ bpk_enc,
                 const unsigned short* __restrict__ bpk_dec,
                 const float* __restrict__ bias_enc,
                 const float* __restrict__ bias_dec,
                 float* __restrict__ out) {
    __shared__ __align__(32) unsigned short A_lds[KBLK][32][16]; // WMMA A layout, bf16
    __shared__ float hf_lds[16][256];                            // f32 h (decoder pred)
    __shared__ float tok_lds[16][2];

    const int tid  = threadIdx.x;
    const int lane = tid & 31;
    const int wave = tid >> 5;
    const int lm   = lane & 15;
    const int lhi  = lane >> 4;
    const int b0   = blockIdx.x * 16;   // 16 batch rows per block

    // ---- init: zero A (h0 = 0 + zero K-padding) ---------------------------
    for (int i = tid; i < KBLK * 32 * 16; i += 256)
        ((unsigned short*)A_lds)[i] = 0;

    // ---- per-lane bias registers (4 gates x 2 column-pairs, enc + dec) ----
    float bE[2][4], bD[2][4];
#pragma unroll
    for (int p = 0; p < 2; ++p) {
        int j = ((2 * wave + p) << 4) + lm;
#pragma unroll
        for (int g = 0; g < 4; ++g) {
            bE[p][g] = bias_enc[g * 256 + j];
            bD[p][g] = bias_dec[g * 256 + j];
        }
    }

    float creg[16];
#pragma unroll
    for (int q = 0; q < 16; ++q) creg[q] = 0.0f;

    __syncthreads();

    // z = [x_t | h] @ B, accumulated per-gate in registers.
    // Launder a loop-variant *integer* offset (keeps the pointer's global
    // address-space provenance -> global_load, not flat_load), fold lane/wave
    // into one base so all 72 tile loads use compile-time immediate offsets.
#define WMMA_STEP(BPK)                                                         \
    do {                                                                       \
        size_t off_ = 0;                                                       \
        asm volatile("" : "+s"(off_)); /* defeat LICM: B streams from L2 */    \
        const unsigned short* bp_ = (BPK) + off_ + ((size_t)lane << 4) +       \
                                    ((size_t)(2 * wave) * TILE_ELEMS);         \
        _Pragma("unroll")                                                      \
        for (int kb = 0; kb < KBLK; ++kb) {                                    \
            bf16x16 a = *(const bf16x16*)&A_lds[kb][lane][0];                  \
            _Pragma("unroll")                                                  \
            for (int p = 0; p < 2; ++p) {                                      \
                const unsigned short* tb = bp_ +                               \
                    (size_t)(kb * NTILES + p) * TILE_ELEMS;                    \
                bf16x16 m0 = *(const bf16x16*)(tb);                            \
                bf16x16 m1 = *(const bf16x16*)(tb + 16 * TILE_ELEMS);          \
                bf16x16 m2 = *(const bf16x16*)(tb + 32 * TILE_ELEMS);          \
                bf16x16 m3 = *(const bf16x16*)(tb + 48 * TILE_ELEMS);          \
                zi[p] = __builtin_amdgcn_wmma_f32_16x16x32_bf16(               \
                    false, a, false, m0, (short)0, zi[p], false, false);       \
                zf[p] = __builtin_amdgcn_wmma_f32_16x16x32_bf16(               \
                    false, a, false, m1, (short)0, zf[p], false, false);       \
                zg[p] = __builtin_amdgcn_wmma_f32_16x16x32_bf16(               \
                    false, a, false, m2, (short)0, zg[p], false, false);       \
                zo[p] = __builtin_amdgcn_wmma_f32_16x16x32_bf16(               \
                    false, a, false, m3, (short)0, zo[p], false, false);       \
            }                                                                  \
        }                                                                      \
    } while (0)

    // gate math straight out of accumulators; h -> A_lds (WMMA A layout)
#define GATE_MATH(BREG, WRITE_HF)                                              \
    do {                                                                       \
        _Pragma("unroll")                                                      \
        for (int p = 0; p < 2; ++p) {                                          \
            int j  = ((2 * wave + p) << 4) + lm;                               \
            int K  = 8 + j;                                                    \
            int kb = K >> 5;                                                   \
            int kk = K & 31;                                                   \
            int lt = (kk >> 4) << 4;                                           \
            int e  = kk & 15;                                                  \
            _Pragma("unroll")                                                  \
            for (int r = 0; r < 8; ++r) {                                      \
                float gi = sigm(zi[p][r] + BREG[p][0]);                        \
                float gf = sigm(zf[p][r] + BREG[p][1]);                        \
                float gg = tanh_f(zg[p][r] + BREG[p][2]);                      \
                float go = sigm(zo[p][r] + BREG[p][3]);                        \
                int   q  = p * 8 + r;                                          \
                float cn = gf * creg[q] + gi * gg;                             \
                creg[q]  = cn;                                                 \
                float hn = go * tanh_f(cn);                                    \
                int   m  = r + (lhi << 3);                                     \
                A_lds[kb][m + lt][e] = f2bf(hn);                               \
                if (WRITE_HF) hf_lds[m][j] = hn;                               \
            }                                                                  \
        }                                                                      \
    } while (0)

    // ======================= encoder: 1024 steps ===========================
    if (tid < 128) {  // input slots K=0..7 for t=0
        int m = tid >> 3, f = tid & 7;
        A_lds[0][m][f] = f2bf(pulse[((size_t)(b0 + m) * T_STEPS) * 8 + f]);
    }

    for (int t = 0; t < T_STEPS; ++t) {
        __syncthreads();                 // A complete (h from prev, input slot)
        f32x8 zi[2] = {}, zf[2] = {}, zg[2] = {}, zo[2] = {};
        WMMA_STEP(bpk_enc);
        __syncthreads();                 // all reads of A done
        GATE_MATH(bE, 0);                // h(t) -> A
        if (tid < 128 && t + 1 < T_STEPS) {  // stage input for t+1
            int m = tid >> 3, f = tid & 7;
            float v = pulse[((size_t)(b0 + m) * T_STEPS + (t + 1)) * 8 + f];
            A_lds[0][m][f] = f2bf(v);
        }
    }

    // ======================= decoder: 35 steps =============================
    if (tid < 32) tok_lds[tid >> 1][tid & 1] = embed_table[tid & 1];
    __syncthreads();

    for (int s = 0; s < DEC_STEPS; ++s) {
        if (tid < 128) {  // stage tok into A slots K=0..1, zero K=2..7
            int m = tid >> 3, f = tid & 7;
            float v = (f < 2) ? tok_lds[m][f] : 0.0f;
            A_lds[0][m][f] = f2bf(v);
        }
        __syncthreads();                 // A complete
        f32x8 zi[2] = {}, zf[2] = {}, zg[2] = {}, zo[2] = {};
        WMMA_STEP(bpk_dec);
        __syncthreads();                 // A reads done
        GATE_MATH(bD, 1);                // h -> A, hf
        __syncthreads();                 // hf complete
        if (tid < 32) {   // pred = h @ W_mlp + b_mlp ; out = pred*sw + sb
            int m = tid >> 1, e = tid & 1;
            float acc = b_mlp[e];
            for (int j = 0; j < 256; ++j)
                acc += hf_lds[m][j] * W_mlp[j * 2 + e];
            tok_lds[m][e] = acc;  // raw pred recurses
            out[((size_t)(b0 + m) * DEC_STEPS + s) * 2 + e] =
                acc * scale_w[e] + scale_b[e];
        }
        __syncthreads();                 // tok ready for next fill
    }
#undef WMMA_STEP
#undef GATE_MATH
}

// ---------------------------------------------------------------------------
extern "C" void kernel_launch(void* const* d_in, const int* in_sizes, int n_in,
                              void* d_out, int out_size, void* d_ws, size_t ws_size,
                              hipStream_t stream) {
    (void)in_sizes; (void)n_in; (void)out_size; (void)ws_size;
    const float* pulse       = (const float*)d_in[0];
    const float* bn_gamma    = (const float*)d_in[1];
    const float* bn_beta     = (const float*)d_in[2];
    const float* bn_mean     = (const float*)d_in[3];
    const float* bn_var      = (const float*)d_in[4];
    const float* W_pulse     = (const float*)d_in[5];
    const float* b_pulse     = (const float*)d_in[6];
    const float* lstm_k      = (const float*)d_in[7];
    const float* lstm_rk     = (const float*)d_in[8];
    const float* lstm_b      = (const float*)d_in[9];
    const float* embed_table = (const float*)d_in[10];
    const float* W_eis       = (const float*)d_in[11];
    const float* b_eis       = (const float*)d_in[12];
    const float* W_mlp       = (const float*)d_in[13];
    const float* b_mlp       = (const float*)d_in[14];
    const float* scale_w     = (const float*)d_in[15];
    const float* scale_b     = (const float*)d_in[16];
    float* out = (float*)d_out;

    const size_t BPK_ELEMS = (size_t)KBLK * NTILES * 32 * 16;  // per matrix (ushort)
    unsigned short* bpk_enc = (unsigned short*)d_ws;
    unsigned short* bpk_dec = bpk_enc + BPK_ELEMS;
    float* bias_enc = (float*)((char*)d_ws + 2 * BPK_ELEMS * sizeof(unsigned short));
    float* bias_dec = bias_enc + NDIM;

    {   // fold + pack weights into WMMA B layout (bf16)
        int total = 2 * KDIM * NDIM;
        pack_B_kernel<<<(total + 255) / 256, 256, 0, stream>>>(
            W_pulse, W_eis, lstm_k, lstm_rk, bn_gamma, bn_var, bpk_enc, bpk_dec);
    }
    {   // folded biases
        bias_kernel<<<8, 256, 0, stream>>>(
            bn_gamma, bn_beta, bn_mean, bn_var, W_pulse, b_pulse,
            lstm_k, lstm_b, b_eis, bias_enc, bias_dec);
    }
    // persistent recurrent kernel: 16 blocks (one M-tile of 16 batch rows each)
    lstm_kernel<<<16, 256, 0, stream>>>(
        pulse, embed_table, W_mlp, b_mlp, scale_w, scale_b,
        bpk_enc, bpk_dec, bias_enc, bias_dec, out);
}
